// MoELayer_3719441678848
// MI455X (gfx1250) — matrix operand order repaired and checked by reading, not compiled, not run on hardware
//
#include <hip/hip_runtime.h>
#include <hip/hip_bf16.h>

#define T_TOK 16384
#define H_DIM 1024
#define F_DIM 4096
#define E_NUM 8
#define MT    32                    // token rows per block tile (2 x 16-row groups)
#define FC    128                   // F-chunk per outer iteration (8 waves x 16 cols)
#define XS_STRIDE (H_DIM + 8)       // bf16 elems; +8 keeps ds_load_b128 bank-spread
#define HS_STRIDE (FC + 8)

typedef __attribute__((ext_vector_type(16))) __bf16 v16bf;
typedef __attribute__((ext_vector_type(8)))  float  v8f;

union Frag { v16bf v; uint4 q[2]; unsigned int u[8]; };

__device__ __forceinline__ unsigned short f2bf(float f) {
  unsigned int u = __float_as_uint(f);
  u += 0x7FFFu + ((u >> 16) & 1u);
  return (unsigned short)(u >> 16);
}
__device__ __forceinline__ unsigned int pack2bf(float lo, float hi) {
#if __has_builtin(__builtin_amdgcn_cvt_pk_bf16_f32)
  typedef __bf16 bf2 __attribute__((ext_vector_type(2)));
  bf2 r = __builtin_amdgcn_cvt_pk_bf16_f32(lo, hi);
  return __builtin_bit_cast(unsigned int, r);
#else
  return (unsigned int)f2bf(lo) | ((unsigned int)f2bf(hi) << 16);
#endif
}

// ---------------------------------------------------------------------------
// Router: logits = x @ Wg, top-2, renormalized gates, append to expert lists.
// ---------------------------------------------------------------------------
__global__ __launch_bounds__(256) void moe_router(
    const float* __restrict__ x, const float* __restrict__ Wg,
    int* __restrict__ cnt, int* __restrict__ lists, float* __restrict__ gates)
{
  const int wave = threadIdx.x >> 5;
  const int lane = threadIdx.x & 31;
  const int t = blockIdx.x * 8 + wave;

  float acc[E_NUM];
#pragma unroll
  for (int e = 0; e < E_NUM; ++e) acc[e] = 0.f;

  const float* xr = x + (size_t)t * H_DIM;
  for (int i = lane; i < H_DIM; i += 32) {
    float xv = xr[i];
    const float4* wr = (const float4*)(Wg + (size_t)i * E_NUM);
    float4 w0 = wr[0], w1 = wr[1];
    acc[0] += xv * w0.x; acc[1] += xv * w0.y;
    acc[2] += xv * w0.z; acc[3] += xv * w0.w;
    acc[4] += xv * w1.x; acc[5] += xv * w1.y;
    acc[6] += xv * w1.z; acc[7] += xv * w1.w;
  }
#pragma unroll
  for (int e = 0; e < E_NUM; ++e) {
    float v = acc[e];
    v += __shfl_xor(v, 16, 32);
    v += __shfl_xor(v, 8, 32);
    v += __shfl_xor(v, 4, 32);
    v += __shfl_xor(v, 2, 32);
    v += __shfl_xor(v, 1, 32);
    acc[e] = v;
  }
  if (lane == 0) {
    int e1 = 0; float l1 = acc[0];
#pragma unroll
    for (int e = 1; e < E_NUM; ++e) if (acc[e] > l1) { l1 = acc[e]; e1 = e; }
    int e2 = 0; float l2 = -3.4e38f;
#pragma unroll
    for (int e = 0; e < E_NUM; ++e)
      if (e != e1 && acc[e] > l2) { l2 = acc[e]; e2 = e; }
    float p1 = 1.f / (1.f + __expf(l2 - l1));   // renormalized top-2 softmax
    float p2 = 1.f - p1;
    int pos1 = atomicAdd(&cnt[e1], 1);
    lists[e1 * T_TOK + pos1] = t;
    gates[e1 * T_TOK + pos1] = p1;
    int pos2 = atomicAdd(&cnt[e2], 1);
    lists[e2 * T_TOK + pos2] = t;
    gates[e2 * T_TOK + pos2] = p2;
  }
}

// ---------------------------------------------------------------------------
// Weight pre-pack: fp32 -> bf16 in exact WMMA B-fragment order.
// One wave per 32(K)x16(N) tile; lane stores its 8 u32 contiguously (2xb128).
// ---------------------------------------------------------------------------
__global__ __launch_bounds__(256) void pack_w1(const float* __restrict__ w,
                                               unsigned short* __restrict__ out)
{
  int tile = blockIdx.x * 8 + (threadIdx.x >> 5);   // (e*256 + ft)*32 + kt
  int lane = threadIdx.x & 31;
  int kt = tile & 31, rest = tile >> 5;
  int ft = rest & 255, e = rest >> 8;
  const float* src = w + (size_t)e * H_DIM * F_DIM
                       + (size_t)(kt * 32 + (lane >> 4) * 16) * F_DIM
                       + ft * 16 + (lane & 15);
  unsigned int u[8];
#pragma unroll
  for (int j = 0; j < 8; ++j)
    u[j] = pack2bf(src[(size_t)(2 * j) * F_DIM], src[(size_t)(2 * j + 1) * F_DIM]);
  uint4* dst = (uint4*)(out + ((size_t)tile << 9) + lane * 16);
  dst[0] = make_uint4(u[0], u[1], u[2], u[3]);
  dst[1] = make_uint4(u[4], u[5], u[6], u[7]);
}

__global__ __launch_bounds__(256) void pack_w2(const float* __restrict__ w,
                                               unsigned short* __restrict__ out)
{
  int tile = blockIdx.x * 8 + (threadIdx.x >> 5);   // (e*128 + fk)*64 + ht
  int lane = threadIdx.x & 31;
  int ht = tile & 63, rest = tile >> 6;
  int fk = rest & 127, e = rest >> 7;
  const float* src = w + (size_t)e * F_DIM * H_DIM
                       + (size_t)(fk * 32 + (lane >> 4) * 16) * H_DIM
                       + ht * 16 + (lane & 15);
  unsigned int u[8];
#pragma unroll
  for (int j = 0; j < 8; ++j)
    u[j] = pack2bf(src[(size_t)(2 * j) * H_DIM], src[(size_t)(2 * j + 1) * H_DIM]);
  uint4* dst = (uint4*)(out + ((size_t)tile << 9) + lane * 16);
  dst[0] = make_uint4(u[0], u[1], u[2], u[3]);
  dst[1] = make_uint4(u[4], u[5], u[6], u[7]);
}

__global__ __launch_bounds__(256) void pack_x(const float* __restrict__ x,
                                              unsigned short* __restrict__ xb)
{
  size_t i = (size_t)blockIdx.x * 256 + threadIdx.x;  // 4 elements each
  float4 v = ((const float4*)x)[i];
  uint2 p; p.x = pack2bf(v.x, v.y); p.y = pack2bf(v.z, v.w);
  ((uint2*)xb)[i] = p;
}

// ---------------------------------------------------------------------------
// Fused expert FFN. MT=32 tokens/block: two 16-row A groups share every B
// fragment (halves L2 weight traffic per FLOP, 2 independent WMMAs per wait).
// Dynamic LDS (~75 KB) — gfx1250 WGP has 320 KB.
// ---------------------------------------------------------------------------
template <bool PACKED>
__global__ __launch_bounds__(256) void moe_expert_ffn(
    const float* __restrict__ x, const unsigned short* __restrict__ xb,
    const float* __restrict__ w1, const float* __restrict__ w2,
    const unsigned short* __restrict__ w1p, const unsigned short* __restrict__ w2p,
    const int* __restrict__ cnt, const int* __restrict__ lists,
    const float* __restrict__ gates, float* __restrict__ y)
{
  extern __shared__ __align__(16) char smem[];
  unsigned short* xs = (unsigned short*)smem;                              // MT*XS_STRIDE
  unsigned short* hs = (unsigned short*)(smem + (size_t)MT * XS_STRIDE * 2);
  float* sgate = (float*)(smem + (size_t)MT * XS_STRIDE * 2 + (size_t)MT * HS_STRIDE * 2);
  int*   sidx  = (int*)(sgate + MT);

  const int e     = blockIdx.y;
  const int count = cnt[e];
  const int m0    = blockIdx.x * MT;
  if (m0 >= count) return;

  if (threadIdx.x < MT) {
    int g  = m0 + threadIdx.x;
    int ok = g < count;
    sidx[threadIdx.x]  = ok ? lists[e * T_TOK + g] : 0;
    sgate[threadIdx.x] = ok ? gates[e * T_TOK + g] : 0.f;
  }
  __syncthreads();

  if constexpr (PACKED) {
    for (int i = threadIdx.x; i < MT * (H_DIM / 8); i += 256) {  // 128 uint4/row
      int m = i >> 7, c = i & 127;
      uint4 v = ((const uint4*)(xb + (size_t)sidx[m] * H_DIM))[c];
      *(uint4*)(&xs[m * XS_STRIDE + c * 8]) = v;
    }
  } else {
    for (int i = threadIdx.x; i < MT * (H_DIM / 4); i += 256) {
      int m = i >> 8, c4 = i & 255;
      float4 xv = ((const float4*)(x + (size_t)sidx[m] * H_DIM))[c4];
      uint2 p; p.x = pack2bf(xv.x, xv.y); p.y = pack2bf(xv.z, xv.w);
      *(uint2*)(&xs[m * XS_STRIDE + c4 * 4]) = p;
    }
  }
  __syncthreads();

  const int wv    = threadIdx.x >> 5;
  const int lane  = threadIdx.x & 31;
  const int mrow  = lane & 15;
  const int khalf = lane >> 4;

  const float* w1e = w1 + (size_t)e * H_DIM * F_DIM;
  const float* w2e = w2 + (size_t)e * F_DIM * H_DIM;

  v8f yacc[16] = {};                 // 32 tokens x 128 H-cols per wave

  for (int f0 = 0; f0 < F_DIM; f0 += FC) {
    // ---- GEMM1: h tiles (2 x 16 tokens) x 16 cols at f0 + wv*16 ----
    v8f hacc0 = {}, hacc1 = {};
    const int ncol = f0 + wv * 16 + mrow;
    const unsigned short* bp =
        w1p + (((size_t)(e * (F_DIM / 16) + (f0 >> 4) + wv)) << 14) + lane * 16;
    for (int k0 = 0; k0 < H_DIM; k0 += 32, bp += 512) {
      Frag a0, a1, b;
      const unsigned short* xr0 = &xs[mrow * XS_STRIDE + k0 + khalf * 16];
      const unsigned short* xr1 = &xs[(16 + mrow) * XS_STRIDE + k0 + khalf * 16];
      a0.q[0] = *(const uint4*)(xr0); a0.q[1] = *(const uint4*)(xr0 + 8);
      a1.q[0] = *(const uint4*)(xr1); a1.q[1] = *(const uint4*)(xr1 + 8);
      if constexpr (PACKED) {
        __builtin_prefetch(bp + 2048, 0, 1);          // 4 tiles ahead
        b.q[0] = *(const uint4*)(bp);
        b.q[1] = *(const uint4*)(bp + 8);
      } else {
        const float* wp = w1e + (size_t)(k0 + khalf * 16) * F_DIM + ncol;
#pragma unroll
        for (int j = 0; j < 8; ++j)
          b.u[j] = pack2bf(wp[(size_t)(2 * j) * F_DIM],
                           wp[(size_t)(2 * j + 1) * F_DIM]);
      }
      hacc0 = __builtin_amdgcn_wmma_f32_16x16x32_bf16(false, a0.v, false, b.v,
                                                      (short)0, hacc0, false, false);
      hacc1 = __builtin_amdgcn_wmma_f32_16x16x32_bf16(false, a1.v, false, b.v,
                                                      (short)0, hacc1, false, false);
    }
    // silu * gate -> hs (fast sigmoid: v * v_rcp(1+exp(-v)))
#pragma unroll
    for (int r = 0; r < 8; ++r) {
      int M = r + khalf * 8;
      float v0 = hacc0[r];
      float s0 = v0 * __builtin_amdgcn_rcpf(1.f + __expf(-v0)) * sgate[M];
      hs[M * HS_STRIDE + wv * 16 + mrow] = f2bf(s0);
      float v1 = hacc1[r];
      float s1 = v1 * __builtin_amdgcn_rcpf(1.f + __expf(-v1)) * sgate[16 + M];
      hs[(16 + M) * HS_STRIDE + wv * 16 + mrow] = f2bf(s1);
    }
    __syncthreads();

    // ---- GEMM2: yacc[32 x (wv*128..+128)] += h_chunk @ w2[f0:f0+FC, :] ----
#pragma unroll
    for (int kk = 0; kk < FC; kk += 32) {
      Frag a20, a21;
      const unsigned short* hr0 = &hs[mrow * HS_STRIDE + kk + khalf * 16];
      const unsigned short* hr1 = &hs[(16 + mrow) * HS_STRIDE + kk + khalf * 16];
      a20.q[0] = *(const uint4*)(hr0); a20.q[1] = *(const uint4*)(hr0 + 8);
      a21.q[0] = *(const uint4*)(hr1); a21.q[1] = *(const uint4*)(hr1 + 8);
      const unsigned short* b2b =
          w2p + (((size_t)((e * (F_DIM / 32) + ((f0 + kk) >> 5)) * (H_DIM / 16)
                           + wv * 8)) << 9) + lane * 16;
#pragma unroll
      for (int nt = 0; nt < 8; ++nt) {
        Frag b2;
        if constexpr (PACKED) {
          __builtin_prefetch(b2b + nt * 512 + 32768, 0, 1);   // next kk stripe
          b2.q[0] = *(const uint4*)(b2b + nt * 512);
          b2.q[1] = *(const uint4*)(b2b + nt * 512 + 8);
        } else {
          const int nc = wv * 128 + nt * 16 + mrow;
          const float* wq = w2e + (size_t)(f0 + kk + khalf * 16) * H_DIM + nc;
#pragma unroll
          for (int j = 0; j < 8; ++j)
            b2.u[j] = pack2bf(wq[(size_t)(2 * j) * H_DIM],
                              wq[(size_t)(2 * j + 1) * H_DIM]);
        }
        yacc[nt * 2]     = __builtin_amdgcn_wmma_f32_16x16x32_bf16(
            false, a20.v, false, b2.v, (short)0, yacc[nt * 2], false, false);
        yacc[nt * 2 + 1] = __builtin_amdgcn_wmma_f32_16x16x32_bf16(
            false, a21.v, false, b2.v, (short)0, yacc[nt * 2 + 1], false, false);
      }
    }
    __syncthreads();
  }

  // epilogue: exactly K=2 commutative fp32 atomic adds per y element
#pragma unroll
  for (int nt = 0; nt < 8; ++nt) {
#pragma unroll
    for (int g = 0; g < 2; ++g) {
#pragma unroll
      for (int r = 0; r < 8; ++r) {
        int M   = g * 16 + khalf * 8 + r;
        int col = wv * 128 + nt * 16 + mrow;
        unsafeAtomicAdd(&y[(size_t)sidx[M] * H_DIM + col], yacc[nt * 2 + g][r]);
      }
    }
  }
}

// ---------------------------------------------------------------------------
static constexpr size_t OFF_LISTS = 256;
static constexpr size_t OFF_GATES = OFF_LISTS + (size_t)E_NUM * T_TOK * 4;
static constexpr size_t OFF_XB    = OFF_GATES + (size_t)E_NUM * T_TOK * 4;
static constexpr size_t OFF_W1P   = OFF_XB + (size_t)T_TOK * H_DIM * 2;
static constexpr size_t OFF_W2P   = OFF_W1P + (size_t)E_NUM * H_DIM * F_DIM * 2;
static constexpr size_t WS_NEED   = OFF_W2P + (size_t)E_NUM * F_DIM * H_DIM * 2;

static constexpr unsigned SMEM_FFN =
    MT * XS_STRIDE * 2 + MT * HS_STRIDE * 2 + MT * 4 + MT * 4;   // ~75 KB

extern "C" void kernel_launch(void* const* d_in, const int* in_sizes, int n_in,
                              void* d_out, int out_size, void* d_ws, size_t ws_size,
                              hipStream_t stream) {
  (void)in_sizes; (void)n_in;
  const float* x  = (const float*)d_in[0];
  const float* Wg = (const float*)d_in[1];
  const float* w1 = (const float*)d_in[2];
  const float* w2 = (const float*)d_in[3];
  float* y = (float*)d_out;

  char* ws = (char*)d_ws;
  int*   cnt   = (int*)ws;
  int*   lists = (int*)(ws + OFF_LISTS);
  float* gates = (float*)(ws + OFF_GATES);
  unsigned short* xb  = (unsigned short*)(ws + OFF_XB);
  unsigned short* w1p = (unsigned short*)(ws + OFF_W1P);
  unsigned short* w2p = (unsigned short*)(ws + OFF_W2P);

  hipMemsetAsync(d_out, 0, (size_t)out_size * sizeof(float), stream);
  hipMemsetAsync(cnt, 0, 256, stream);

  moe_router<<<dim3(T_TOK / 8), dim3(256), 0, stream>>>(x, Wg, cnt, lists, gates);

  if (ws_size >= WS_NEED) {
    pack_x<<<dim3(T_TOK * H_DIM / 1024), dim3(256), 0, stream>>>(x, xb);
    pack_w1<<<dim3(E_NUM * (F_DIM / 16) * (H_DIM / 32) / 8), dim3(256), 0, stream>>>(w1, w1p);
    pack_w2<<<dim3(E_NUM * (F_DIM / 32) * (H_DIM / 16) / 8), dim3(256), 0, stream>>>(w2, w2p);
    moe_expert_ffn<true><<<dim3(T_TOK / MT, E_NUM), dim3(256), SMEM_FFN, stream>>>(
        x, xb, w1, w2, w1p, w2p, cnt, lists, gates, y);
  } else {
    moe_expert_ffn<false><<<dim3(T_TOK / MT, E_NUM), dim3(256), SMEM_FFN, stream>>>(
        x, xb, w1, w2, w1p, w2p, cnt, lists, gates, y);
  }
}
